// CrossAttention_53592601919512
// MI455X (gfx1250) — matrix operand order repaired and checked by reading, not compiled, not run on hardware
//
#include <hip/hip_runtime.h>
#include <math.h>

typedef unsigned short u16;
typedef __attribute__((ext_vector_type(16))) __bf16 v16bf;
typedef __attribute__((ext_vector_type(8)))  float  v8f;
typedef __attribute__((ext_vector_type(8)))  u16    us8;
typedef __attribute__((ext_vector_type(4)))  u16    us4;
typedef __attribute__((ext_vector_type(4)))  float  f32x4;

#define CH    128
#define NROW  32768        // B*N = 2*16384
#define ROWS  16           // rows per wave M-tile (matches WMMA M)
#define ASTR  136          // padded bf16 row stride for A staging (bank-spread)
#define WAVES 4
#define NTHR  (WAVES*32)
#define NBLK  256
#define MB_TOT (NROW/ROWS) // 2048

__device__ __forceinline__ u16 f2bf(float f) {
  union { float f; unsigned u; } v; v.f = f;
  unsigned r = v.u + 0x7FFFu + ((v.u >> 16) & 1u);
  return (u16)(r >> 16);
}
__device__ __forceinline__ float bf2f(u16 h) {
  union { unsigned u; float f; } v; v.u = ((unsigned)h) << 16;
  return v.f;
}

// A fragment (16x32 bf16) from padded row-major LDS staging.
__device__ __forceinline__ v16bf load_a_frag(const u16* stage, int kt, int lane) {
  const int m = lane & 15, half = lane >> 4;
  const u16* p = stage + m * ASTR + kt * 32 + half * 8;
  union { us8 h[2]; v16bf v; } u;
  u.h[0] = *(const us8*)(p);
  u.h[1] = *(const us8*)(p + 16);
  return u.v;
}

// B fragment (32x16 bf16) pre-swizzled: 32 bytes contiguous per lane per tile.
__device__ __forceinline__ v16bf load_b_frag(const u16* w, int tile, int lane) {
  return *(const v16bf*)(w + tile * 512 + lane * 16);
}

__device__ __forceinline__ v8f wmma_bf16(v16bf a, v16bf b, v8f c) {
  return __builtin_amdgcn_wmma_f32_16x16x32_bf16(false, a, false, b, (short)0, c, false, false);
}

// Swizzle a KxN fp32 weight (row-major) into WMMA B-fragment order as bf16.
__device__ __forceinline__ void fill_w(u16* dst, const float* W, int Kdim, int Ndim,
                                       int tid, int nth) {
  const int NT = Ndim >> 4;
  const int total = Kdim * Ndim;
  for (int i = tid; i < total; i += nth) {
    int tile = i >> 9;
    int r = i & 511;
    int l = r >> 4;
    int e = r & 15;
    int kt = tile / NT, nt = tile - kt * NT;
    int kk = kt * 32 + ((e >> 3) << 4) + ((l >> 4) << 3) + (e & 7);
    int nn = nt * 16 + (l & 15);
    dst[i] = f2bf(W[kk * Ndim + nn]);
  }
}

// Async copy global -> LDS (no VGPR data path), tracked by ASYNCcnt.
__device__ __forceinline__ void async_copy_b128(unsigned lds_byte_addr,
                                                unsigned gbl_byte_off,
                                                const void* saddr) {
  asm volatile("global_load_async_to_lds_b128 %0, %1, %2"
               :: "v"(lds_byte_addr), "v"(gbl_byte_off), "s"(saddr)
               : "memory");
}
__device__ __forceinline__ void wait_async0() {
  asm volatile("s_wait_asynccnt 0" ::: "memory");
}

// Async-stage one 16x128 f32 tile (8 KB) into contiguous LDS: 16 b128 ops/lane.
__device__ __forceinline__ void stage_f32_tile_async(float* dst, const float* src, int lane) {
#pragma unroll
  for (int it = 0; it < 16; ++it) {
    int i = lane + it * 32;                 // 16-byte chunk index
    unsigned ldsa = (unsigned)(unsigned long long)(dst + i * 4);
    async_copy_b128(ldsa, (unsigned)(i * 16), (const void*)src);
  }
}

// ---------------------------------------------------------------------------
// Fully fused transformer block: all 5 weight matrices resident in LDS (bf16),
// one 16-row M-tile per wave per iteration, ~310 KB LDS per workgroup.
// ---------------------------------------------------------------------------
__launch_bounds__(NTHR, 1)
__global__ void fused_block(const float* __restrict__ query_in,
                            const float* __restrict__ kv_in,
                            const float* __restrict__ w_kv,
                            const float* __restrict__ w_q,
                            const float* __restrict__ w_mh,
                            const float* __restrict__ w1w,
                            const float* __restrict__ w2w,
                            const float* __restrict__ b_mh,
                            const float* __restrict__ b1,
                            const float* __restrict__ b2,
                            const float* __restrict__ ln1_g,
                            const float* __restrict__ ln1_b,
                            const float* __restrict__ ln2_g,
                            const float* __restrict__ ln2_b,
                            float* __restrict__ out) {
  // weights (192 KB) + params (3.5 KB)
  __shared__ __align__(32) u16 lds_wq [128 * 128];
  __shared__ __align__(32) u16 lds_wkv[128 * 256];
  __shared__ __align__(32) u16 lds_wmh[128 * 128];
  __shared__ __align__(32) u16 lds_w1 [128 * 128];
  __shared__ __align__(32) u16 lds_w2 [128 * 128];
  __shared__ float lds_p[7 * 128];  // b_mh,b1,b2,ln1_g,ln1_b,ln2_g,ln2_b
  // per-wave working set (28.5 KB each), heavily reused across phases:
  __shared__ __align__(16) u16   lds_qa [WAVES][ROWS * ASTR]; // query A-stage -> attn-out A-stage
  __shared__ __align__(16) u16   lds_qo [WAVES][ROWS * 128];  // q (bf16)
  __shared__ __align__(16) u16   lds_kva[WAVES][ROWS * ASTR]; // kv A-stage -> res/gelu bf16 stage
  __shared__ __align__(16) u16   lds_kvo[WAVES][ROWS * 256];  // k|v (bf16) -> f32 acc dump (cast)
  __shared__ __align__(16) float lds_y  [WAVES][ROWS * 128];  // async query f32 -> residual (in place)

  const int tid = threadIdx.x;
  const int lane = tid & 31;
  const int w = tid >> 5;

  fill_w(lds_wq, w_q, 128, 128, tid, NTHR);
  fill_w(lds_wkv, w_kv, 128, 256, tid, NTHR);
  fill_w(lds_wmh, w_mh, 128, 128, tid, NTHR);
  fill_w(lds_w1, w1w, 128, 128, tid, NTHR);
  fill_w(lds_w2, w2w, 128, 128, tid, NTHR);
  {
    const float* srcs[7] = {b_mh, b1, b2, ln1_g, ln1_b, ln2_g, ln2_b};
    for (int i = tid; i < 7 * 128; i += NTHR) lds_p[i] = srcs[i >> 7][i & 127];
  }
  __syncthreads();

  u16* qa = lds_qa[w];          // phase 1-2: query A-staging
  u16* ao = lds_qa[w];          // phase 3-4: attention-out A-staging (same region)
  u16* qo = lds_qo[w];
  u16* ka = lds_kva[w];         // phase 3: kv A-staging; phase 5-7: res/gelu bf16 staging
  u16* ko = lds_kvo[w];         // phase 3: k|v bf16
  float* xx = (float*)lds_kvo[w]; // phase 4-8: f32 acc dump (ko region, 8 KB)
  float* yy = lds_y[w];         // async query f32, then residual in place
  const int mbStride = gridDim.x * WAVES;
  const int mb0 = blockIdx.x * WAVES + w;

  // prime: async-load first tile's query rows (f32) for the residual path
  if (mb0 < MB_TOT)
    stage_f32_tile_async(yy, query_in + (long)mb0 * ROWS * CH, lane);

  for (int mb = mb0; mb < MB_TOT; mb += mbStride) {
    const long rowBase = (long)mb * ROWS;

    if (mb + mbStride < MB_TOT) {
      __builtin_prefetch(query_in + (rowBase + (long)mbStride * ROWS) * CH, 0, 1);
      __builtin_prefetch(kv_in + (rowBase + (long)mbStride * ROWS) * 1024, 0, 1);
    }

    // ---- phase 1: stage 16 query rows (fp32 -> bf16, padded) ----
    {
      const float* src = query_in + rowBase * CH;
      for (int i = lane; i < (ROWS * CH) / 4; i += 32) {
        f32x4 v = *(const f32x4*)(src + i * 4);
        int e0 = i * 4;
        int row = e0 >> 7, col = e0 & 127;
        us4 o; o.x = f2bf(v.x); o.y = f2bf(v.y); o.z = f2bf(v.z); o.w = f2bf(v.w);
        *(us4*)(qa + row * ASTR + col) = o;
      }
    }

    // ---- phase 2: Q GEMM (16x128)@(128x128) -> qo (bf16) ----
    {
      v16bf a[4];
#pragma unroll
      for (int kt = 0; kt < 4; ++kt) a[kt] = load_a_frag(qa, kt, lane);
      for (int nt = 0; nt < 8; ++nt) {
        v8f acc = {0.f, 0.f, 0.f, 0.f, 0.f, 0.f, 0.f, 0.f};
#pragma unroll
        for (int kt = 0; kt < 4; ++kt)
          acc = wmma_bf16(a[kt], load_b_frag(lds_wq, kt * 8 + nt, lane), acc);
        const int col = nt * 16 + (lane & 15);
        const int m0 = (lane >> 4) * 8;
#pragma unroll
        for (int e = 0; e < 8; ++e) qo[(m0 + e) * 128 + col] = f2bf(acc[e]);
      }
    }

    // ---- phase 3: KV GEMM + attention, sub-blocks of 2 query rows ----
    for (int sub = 0; sub < 8; ++sub) {
      const float* src = kv_in + (rowBase + sub * 2) * 1024; // 16 contiguous kv rows
      for (int i = lane; i < (16 * 128) / 4; i += 32) {
        f32x4 v = *(const f32x4*)(src + i * 4);
        int e0 = i * 4;
        int row = e0 >> 7, col = e0 & 127;
        us4 o; o.x = f2bf(v.x); o.y = f2bf(v.y); o.z = f2bf(v.z); o.w = f2bf(v.w);
        *(us4*)(ka + row * ASTR + col) = o;
      }

      v16bf a[4];
#pragma unroll
      for (int kt = 0; kt < 4; ++kt) a[kt] = load_a_frag(ka, kt, lane);
      for (int nt = 0; nt < 16; ++nt) {
        v8f acc = {0.f, 0.f, 0.f, 0.f, 0.f, 0.f, 0.f, 0.f};
#pragma unroll
        for (int kt = 0; kt < 4; ++kt)
          acc = wmma_bf16(a[kt], load_b_frag(lds_wkv, kt * 16 + nt, lane), acc);
        const int col = nt * 16 + (lane & 15);
        const int m0 = (lane >> 4) * 8;
#pragma unroll
        for (int e = 0; e < 8; ++e) ko[(m0 + e) * 256 + col] = f2bf(acc[e]);
      }

      // attention on all 32 lanes: lane = (row,head) pair x 8-dim chunk
      {
        const int pair = lane >> 2;
        const int chunk = lane & 3;
        const int sr = pair >> 2;
        const int h = pair & 3;
        const int dbase = h * 32 + chunk * 8;

        us8 qraw = *(const us8*)(qo + (sub * 2 + sr) * 128 + dbase);
        float qv[8];
#pragma unroll
        for (int d = 0; d < 8; ++d) qv[d] = bf2f(qraw[d]);

        float sim[8];
        float mx = -1e30f;
#pragma unroll
        for (int k = 0; k < 8; ++k) {
          us8 kraw = *(const us8*)(ko + (sr * 8 + k) * 256 + dbase);
          float s = 0.f;
#pragma unroll
          for (int d = 0; d < 8; ++d) s += qv[d] * bf2f(kraw[d]);
          s += __shfl_xor(s, 1, 32);
          s += __shfl_xor(s, 2, 32);
          s *= 0.17677669529663687f;  // 1/sqrt(32)
          sim[k] = s;
          mx = fmaxf(mx, s);
        }
        float sum = 0.f;
#pragma unroll
        for (int k = 0; k < 8; ++k) { sim[k] = __expf(sim[k] - mx); sum += sim[k]; }
        const float inv = 1.f / sum;

        float o[8];
#pragma unroll
        for (int d = 0; d < 8; ++d) o[d] = 0.f;
#pragma unroll
        for (int k = 0; k < 8; ++k) {
          const float aw = sim[k] * inv;
          us8 vraw = *(const us8*)(ko + (sr * 8 + k) * 256 + 128 + dbase);
#pragma unroll
          for (int d = 0; d < 8; ++d) o[d] += aw * bf2f(vraw[d]);
        }
        us8 ov;
#pragma unroll
        for (int d = 0; d < 8; ++d) ov[d] = f2bf(o[d]);
        *(us8*)(ao + (sub * 2 + sr) * ASTR + dbase) = ov;  // A-staging layout
      }
    }

    // ---- phase 4: GEMM attn@w_mh + b_mh -> xx (f32, reuses ko region) ----
    {
      v16bf a[4];
#pragma unroll
      for (int kt = 0; kt < 4; ++kt) a[kt] = load_a_frag(ao, kt, lane);
      for (int nt = 0; nt < 8; ++nt) {
        v8f acc = {0.f, 0.f, 0.f, 0.f, 0.f, 0.f, 0.f, 0.f};
#pragma unroll
        for (int kt = 0; kt < 4; ++kt)
          acc = wmma_bf16(a[kt], load_b_frag(lds_wmh, kt * 8 + nt, lane), acc);
        const int col = nt * 16 + (lane & 15);
        const int m0 = (lane >> 4) * 8;
        const float bb = lds_p[col];
#pragma unroll
        for (int e = 0; e < 8; ++e) xx[(m0 + e) * 128 + col] = acc[e] + bb;
      }
    }

    // ---- phase 5: ln1 + residual (async query tile, updated in place) ----
    wait_async0();   // yy now holds this tile's query rows (f32)
    {
      const int row = lane & 15;
      const int half = lane >> 4;
      const int c0 = half * 64;
      const float* xr = xx + row * 128;
      float s = 0.f, ss = 0.f;
      for (int c = c0; c < c0 + 64; c += 4) {
        f32x4 v = *(const f32x4*)(xr + c);
        s += v.x + v.y + v.z + v.w;
        ss += v.x * v.x + v.y * v.y + v.z * v.z + v.w * v.w;
      }
      s += __shfl_xor(s, 16, 32);
      ss += __shfl_xor(ss, 16, 32);
      const float mu = s * (1.f / 128.f);
      const float var = ss * (1.f / 128.f) - mu * mu;
      const float rstd = rsqrtf(var + 1e-5f);
      float* yr = yy + row * 128;
      u16* zr = ka + row * ASTR;  // res bf16 staging (kva region, now free)
      for (int c = c0; c < c0 + 64; ++c) {
        float v = (xr[c] - mu) * rstd * lds_p[384 + c] + lds_p[512 + c] + yr[c];
        yr[c] = v;          // residual, in place over query f32
        zr[c] = f2bf(v);
      }
    }

    // ---- phase 6: GEMM res@w1 + b1, exact gelu -> ka (bf16, in place) ----
    {
      v16bf a[4];
#pragma unroll
      for (int kt = 0; kt < 4; ++kt) a[kt] = load_a_frag(ka, kt, lane);
      for (int nt = 0; nt < 8; ++nt) {
        v8f acc = {0.f, 0.f, 0.f, 0.f, 0.f, 0.f, 0.f, 0.f};
#pragma unroll
        for (int kt = 0; kt < 4; ++kt)
          acc = wmma_bf16(a[kt], load_b_frag(lds_w1, kt * 8 + nt, lane), acc);
        const int col = nt * 16 + (lane & 15);
        const int m0 = (lane >> 4) * 8;
        const float bb = lds_p[128 + col];
#pragma unroll
        for (int e = 0; e < 8; ++e) {
          float x = acc[e] + bb;
          float g = 0.5f * x * (1.0f + erff(x * 0.70710678118654752f));
          ka[(m0 + e) * ASTR + col] = f2bf(g);
        }
      }
    }

    // ---- phase 7: GEMM h@w2 + b2 -> xx (f32) ----
    {
      v16bf a[4];
#pragma unroll
      for (int kt = 0; kt < 4; ++kt) a[kt] = load_a_frag(ka, kt, lane);
      for (int nt = 0; nt < 8; ++nt) {
        v8f acc = {0.f, 0.f, 0.f, 0.f, 0.f, 0.f, 0.f, 0.f};
#pragma unroll
        for (int kt = 0; kt < 4; ++kt)
          acc = wmma_bf16(a[kt], load_b_frag(lds_w2, kt * 8 + nt, lane), acc);
        const int col = nt * 16 + (lane & 15);
        const int m0 = (lane >> 4) * 8;
        const float bb = lds_p[256 + col];
#pragma unroll
        for (int e = 0; e < 8; ++e) xx[(m0 + e) * 128 + col] = acc[e] + bb;
      }
    }

    // ---- phase 8: ln2(mlp + res) -> global out, 2 lanes per row ----
    {
      const int row = lane & 15;
      const int half = lane >> 4;
      const int c0 = half * 64;
      const float* xr = xx + row * 128;
      const float* yr = yy + row * 128;
      float s = 0.f, ss = 0.f;
      for (int c = c0; c < c0 + 64; ++c) {
        float t = xr[c] + yr[c];
        s += t;
        ss += t * t;
      }
      s += __shfl_xor(s, 16, 32);
      ss += __shfl_xor(ss, 16, 32);
      const float mu = s * (1.f / 128.f);
      const float var = ss * (1.f / 128.f) - mu * mu;
      const float rstd = rsqrtf(var + 1e-5f);
      float* op = out + (rowBase + row) * CH;
      for (int c = c0; c < c0 + 64; c += 4) {
        f32x4 o;
        o.x = ((xr[c + 0] + yr[c + 0]) - mu) * rstd * lds_p[640 + c + 0] + lds_p[768 + c + 0];
        o.y = ((xr[c + 1] + yr[c + 1]) - mu) * rstd * lds_p[640 + c + 1] + lds_p[768 + c + 1];
        o.z = ((xr[c + 2] + yr[c + 2]) - mu) * rstd * lds_p[640 + c + 2] + lds_p[768 + c + 2];
        o.w = ((xr[c + 3] + yr[c + 3]) - mu) * rstd * lds_p[640 + c + 3] + lds_p[768 + c + 3];
        *(f32x4*)(op + c) = o;
      }
    }

    // ---- issue next tile's async query load (y is free after ln2) ----
    {
      int nmb = (mb + mbStride < MB_TOT) ? (mb + mbStride) : mb; // dummy on last iter
      stage_f32_tile_async(yy, query_in + (long)nmb * ROWS * CH, lane);
    }
  }
}

extern "C" void kernel_launch(void* const* d_in, const int* in_sizes, int n_in,
                              void* d_out, int out_size, void* d_ws, size_t ws_size,
                              hipStream_t stream) {
  const float* query_in = (const float*)d_in[0];
  const float* kv_in    = (const float*)d_in[1];
  const float* w_kv     = (const float*)d_in[2];
  const float* w_q      = (const float*)d_in[3];
  const float* w_mh     = (const float*)d_in[4];
  const float* b_mh     = (const float*)d_in[5];
  const float* w1       = (const float*)d_in[6];
  const float* b1       = (const float*)d_in[7];
  const float* w2       = (const float*)d_in[8];
  const float* b2       = (const float*)d_in[9];
  const float* ln1_g    = (const float*)d_in[10];
  const float* ln1_b    = (const float*)d_in[11];
  const float* ln2_g    = (const float*)d_in[12];
  const float* ln2_b    = (const float*)d_in[13];
  float* outp = (float*)d_out;

  hipLaunchKernelGGL(fused_block, dim3(NBLK), dim3(NTHR), 0, stream,
                     query_in, kv_in, w_kv, w_q, w_mh, w1, w2,
                     b_mh, b1, b2, ln1_g, ln1_b, ln2_g, ln2_b, outp);
}
